// HMemQA_63101659512953
// MI455X (gfx1250) — matrix operand order repaired and checked by reading, not compiled, not run on hardware
//
#include <hip/hip_runtime.h>
#include <cstdint>

// ---------------------------------------------------------------------------
// HMemQA forward for MI455X (gfx1250).
//  - bf16 WMMA (v_wmma_f32_16x16x32_bf16) for all dense GEMMs; waves are
//    register-blocked 4 M-subtiles (64x16 output) so one B fragment feeds
//    4 WMMAs (~1.25KB VMEM per WMMA instead of 2KB).
//  - Hebbian scan: one fused read+write pass over m per step; m (134MB) is
//    L2-resident (192MB global L2); v_h for the NEXT step is accumulated in
//    the same pass (kn = LN(k) is precomputable, independent of m); the
//    LayerNorm of the write pre-activation is fused into the scan kernel.
//  - LDS staging for LN / reductions, global_prefetch_b8 in the scan loop.
// ---------------------------------------------------------------------------

#define B_   128
#define S_   100
#define W_   12
#define E_   256
#define M_   512
#define V_   10000
#define HOPS_ 3
#define GP   0.1f
#define GN   0.1f
#define WMX  1.0f
#define EPS_ 1e-5f

typedef __bf16 bf16_t;
typedef __attribute__((ext_vector_type(16))) __bf16 v16bf;
typedef __attribute__((ext_vector_type(8)))  float  v8f;

// ---------------------------------------------------------------- utilities
__global__ void k_zero(uint32_t* p, int n) {
    int i = blockIdx.x * blockDim.x + threadIdx.x;
    if (i < n) p[i] = 0u;
}

// gather + position-weighted bag of words: out[row,e] = sum_w we[tok,e]*pos[w,e]
__global__ void k_embed(const int* __restrict__ tok, const float* __restrict__ we,
                        const float* __restrict__ pos, float* __restrict__ out) {
    int row = blockIdx.x;
    int e   = threadIdx.x;                 // 256 threads
    float acc = 0.f;
#pragma unroll
    for (int w = 0; w < W_; ++w) {
        int t = tok[row * W_ + w];
        acc = fmaf(we[(size_t)t * E_ + e], pos[w * E_ + e], acc);
    }
    out[(size_t)row * E_ + e] = acc;
}

__global__ void k_stats_story(const float* __restrict__ x, float* sum, float* sq) {
    int e = threadIdx.x, g = blockIdx.x;   // 256 blocks x 256 threads, 50 rows each
    float s1 = 0.f, s2 = 0.f;
    for (int r = g * 50; r < g * 50 + 50; ++r) {
        float v = x[(size_t)r * E_ + e];
        s1 += v; s2 += v * v;
    }
    atomicAdd(&sum[e], s1);
    atomicAdd(&sq[e], s2);
}

__global__ void k_stats_query(const float* __restrict__ x, float* sum, float* sq) {
    int e = threadIdx.x;
    float s1 = 0.f, s2 = 0.f;
    for (int r = 0; r < B_; ++r) {
        float v = x[(size_t)r * E_ + e];
        s1 += v; s2 += v * v;
    }
    sum[e] = s1; sq[e] = s2;
}

__global__ void k_bn_finalize(const float* sum_s, const float* sq_s,
                              const float* sum_q, const float* sq_q,
                              const float* g_s, const float* b_s,
                              const float* g_q, const float* b_q,
                              float* scale_s, float* shift_s,
                              float* scale_q, float* shift_q) {
    int e = threadIdx.x;
    float n1 = (float)(B_ * S_);
    float mu = sum_s[e] / n1;
    float var = sq_s[e] / n1 - mu * mu;
    float sc = g_s[e] * rsqrtf(var + EPS_);
    scale_s[e] = sc; shift_s[e] = b_s[e] - mu * sc;
    float n2 = (float)B_;
    mu  = sum_q[e] / n2;
    var = sq_q[e] / n2 - mu * mu;
    sc  = g_q[e] * rsqrtf(var + EPS_);
    scale_q[e] = sc; shift_q[e] = b_q[e] - mu * sc;
}

__global__ void k_bn_apply_bf16(const float* __restrict__ x, const float* scale,
                                const float* shift, bf16_t* __restrict__ out, int n) {
    int i = blockIdx.x * blockDim.x + threadIdx.x;
    if (i >= n) return;
    int e = i & (E_ - 1);
    out[i] = (bf16_t)(fmaf(x[i], scale[e], shift[e]));
}

__global__ void k_bn_apply_f32(const float* __restrict__ x, const float* scale,
                               const float* shift, float* __restrict__ out, int n) {
    int i = blockIdx.x * blockDim.x + threadIdx.x;
    if (i >= n) return;
    int e = i & (E_ - 1);
    out[i] = fmaf(x[i], scale[e], shift[e]);
}

__global__ void k_cvt_bf16(const float* __restrict__ src, bf16_t* __restrict__ dst,
                           int rows, int cols, int src_ld) {
    int i = blockIdx.x * blockDim.x + threadIdx.x;
    if (i >= rows * cols) return;
    int r = i / cols, c = i - r * cols;
    dst[i] = (bf16_t)src[(size_t)r * src_ld + c];
}

// ----------------------------------------------------------- WMMA bf16 GEMM
// C[MrxN] = A[MrxK] * Bt[NxK]^T (+ bias[N]) (+ Cadd[row*cadd_ld+col]) (relu?)
// Each wave computes a 64x16 output block (4 M-subtiles), reusing one B
// fragment across 4 v_wmma_f32_16x16x32_bf16 issues per K-step.
__global__ __launch_bounds__(256)
void k_gemm_bf16(const bf16_t* __restrict__ A, const bf16_t* __restrict__ Bt,
                 const float* __restrict__ bias, const float* __restrict__ Cadd,
                 int cadd_ld, void* __restrict__ Cout, int out_bf16, int relu,
                 int mtiles4, int ntiles, int K, int ldc) {
    int wave = (blockIdx.x * blockDim.x + threadIdx.x) >> 5;
    if (wave >= mtiles4 * ntiles) return;          // wave-uniform exit
    int lane = threadIdx.x & 31;
    int mt = wave % mtiles4;
    int nt = wave / mtiles4;
    int h   = lane >> 4;       // half-wave select
    int l16 = lane & 15;

    const bf16_t* arow = A  + (size_t)(mt * 64 + l16) * K;
    const bf16_t* brow = Bt + (size_t)(nt * 16 + l16) * K;

    v8f c[4] = {};
    union Frag { v16bf v; uint4 q[2]; };
    for (int k0 = 0; k0 < K; k0 += 32) {
        // B 32x16 bf16 layout: lane column n=l16, K in [k0+h*16, k0+h*16+16)
        Frag bf;
        bf.q[0] = *(const uint4*)(brow + k0 + h * 16);
        bf.q[1] = *(const uint4*)(brow + k0 + h * 16 + 8);
#pragma unroll
        for (int s = 0; s < 4; ++s) {
            // A 16x32 bf16 layout: lane row m=l16, K in [k0+h*16, k0+h*16+16)
            const bf16_t* ar = arow + (size_t)s * 16 * K;
            Frag af;
            af.q[0] = *(const uint4*)(ar + k0 + h * 16);
            af.q[1] = *(const uint4*)(ar + k0 + h * 16 + 8);
            c[s] = __builtin_amdgcn_wmma_f32_16x16x32_bf16(false, af.v, false, bf.v,
                                                           (short)0, c[s], false, false);
        }
    }
    // D 16x16 f32: lane col = l16, rows = r + 8*h for r in 0..7
    int col = nt * 16 + l16;
    float bv = bias ? bias[col] : 0.f;
#pragma unroll
    for (int s = 0; s < 4; ++s) {
        int row0 = mt * 64 + s * 16 + h * 8;
#pragma unroll
        for (int r = 0; r < 8; ++r) {
            int row = row0 + r;
            float val = c[s][r] + bv;
            if (Cadd) val += Cadd[(size_t)row * cadd_ld + col];
            if (relu) val = fmaxf(val, 0.f);
            if (out_bf16) ((bf16_t*)Cout)[(size_t)row * ldc + col] = (bf16_t)val;
            else          ((float*)Cout)[(size_t)row * ldc + col]  = val;
        }
    }
}

// ------------------------------------------------------------- LayerNorm(512)
// permute=1: src row p=(b*S+s) -> dst row (s*B+b)   (kn_all, step-major)
__global__ __launch_bounds__(512)
void k_ln(const float* __restrict__ src, const float* __restrict__ g,
          const float* __restrict__ b, float* __restrict__ dst, int permute) {
    int p = blockIdx.x, t = threadIdx.x;
    float x = src[(size_t)p * M_ + t];
    __shared__ float red[M_];
    red[t] = x; __syncthreads();
    for (int o = M_ / 2; o > 0; o >>= 1) { if (t < o) red[t] += red[t + o]; __syncthreads(); }
    float mu = red[0] * (1.f / M_);
    __syncthreads();
    float d = x - mu;
    red[t] = d * d; __syncthreads();
    for (int o = M_ / 2; o > 0; o >>= 1) { if (t < o) red[t] += red[t + o]; __syncthreads(); }
    float var = red[0] * (1.f / M_);
    size_t drow = permute ? ((size_t)(p % S_) * B_ + (size_t)(p / S_)) : (size_t)p;
    dst[drow * M_ + t] = d * rsqrtf(var + EPS_) * g[t] + b[t];
}

// ------------------------------------------------------ fused Hebbian update
// grid: (b, n-chunk of 64) = B*8 blocks, 256 threads.
// Fuses: v2 = LN2(preact) (redundant per chunk, trivial), then
// m_new = m*(1 - GP*kn*v2 - GN*kn^2) + GP*WMX*kn*v2 ; also accumulates
// v_h_next[n] = sum_m kn_next[m]*m_new[m,n] for the next step (stored bf16).
__global__ __launch_bounds__(256)
void k_hebb(float* __restrict__ mem, const float* __restrict__ kn,
            const float* __restrict__ knx, const float* __restrict__ preact,
            const float* __restrict__ ln2g, const float* __restrict__ ln2b,
            bf16_t* __restrict__ vh16_next, int is_first) {
    int blk = blockIdx.x;
    int b = blk >> 3, cch = blk & 7;
    int n0 = cch * 64;
    int t = threadIdx.x;
    __shared__ float kn_s[M_], knx_s[M_], pre_s[M_], v2_s[64];
    __shared__ float red[256];
    kn_s[t]        = kn[(size_t)b * M_ + t];
    kn_s[t + 256]  = kn[(size_t)b * M_ + t + 256];
    knx_s[t]       = knx[(size_t)b * M_ + t];
    knx_s[t + 256] = knx[(size_t)b * M_ + t + 256];
    pre_s[t]       = preact[(size_t)b * M_ + t];
    pre_s[t + 256] = preact[(size_t)b * M_ + t + 256];
    __syncthreads();

    // LayerNorm(512) of pre_s with 256 threads
    red[t] = pre_s[t] + pre_s[t + 256];
    __syncthreads();
    for (int o = 128; o > 0; o >>= 1) { if (t < o) red[t] += red[t + o]; __syncthreads(); }
    float mu = red[0] * (1.f / M_);
    __syncthreads();
    {
        float d0 = pre_s[t] - mu, d1 = pre_s[t + 256] - mu;
        red[t] = d0 * d0 + d1 * d1;
    }
    __syncthreads();
    for (int o = 128; o > 0; o >>= 1) { if (t < o) red[t] += red[t + o]; __syncthreads(); }
    float rs = rsqrtf(red[0] * (1.f / M_) + EPS_);
    if (t < 64) {
        int n = n0 + t;
        v2_s[t] = (pre_s[n] - mu) * rs * ln2g[n] + ln2b[n];
    }
    __syncthreads();

    int nl = t & 63, g = t >> 6;
    float vv = v2_s[nl];
    float* mp = mem + ((size_t)b * M_ + (size_t)g * 128) * M_ + n0 + nl;
    float acc = 0.f;
#pragma unroll 4
    for (int r = 0; r < 128; ++r) {
        if ((r & 31) == 0) __builtin_prefetch(mp + 32 * M_, 0, 1); // global_prefetch_b8
        float k1   = kn_s[g * 128 + r];
        float a    = GP * k1 * vv;
        float coef = 1.f - a - GN * k1 * k1;
        float mo   = is_first ? 0.f : mp[0];
        float mn   = fmaf(mo, coef, WMX * a);
        mp[0] = mn;
        acc = fmaf(knx_s[g * 128 + r], mn, acc);
        mp += M_;
    }
    __syncthreads();
    red[t] = acc; __syncthreads();
    if (g == 0) {
        float s = red[nl] + red[64 + nl] + red[128 + nl] + red[192 + nl];
        vh16_next[(size_t)b * M_ + n0 + nl] = (bf16_t)s;
    }
}

// ----------------------------------------------------------------- read hops
__global__ __launch_bounds__(512)
void k_read_kr(const float* __restrict__ qe, const float* __restrict__ vr,
               const float* __restrict__ rW, const float* __restrict__ rb,
               float* __restrict__ kr) {
    int b = blockIdx.x, j = threadIdx.x;   // 512 threads
    __shared__ float q_s[E_], v_s[M_];
    v_s[j] = vr[(size_t)b * M_ + j];
    if (j < E_) q_s[j] = qe[(size_t)b * E_ + j];
    __syncthreads();
    const float* wrow = rW + (size_t)j * (E_ + M_);
    float acc = rb[j];
    for (int e = 0; e < E_; ++e) acc = fmaf(q_s[e], wrow[e], acc);
    for (int n = 0; n < M_; ++n) acc = fmaf(v_s[n], wrow[E_ + n], acc);
    kr[(size_t)b * M_ + j] = acc;
}

__global__ __launch_bounds__(256)
void k_read_vr(const float* __restrict__ mem, const float* __restrict__ kr,
               float* __restrict__ vr) {
    int blk = blockIdx.x;
    int b = blk >> 3, c = blk & 7;
    int n0 = c * 64;
    int t = threadIdx.x;
    __shared__ float kr_s[M_];
    kr_s[t]       = kr[(size_t)b * M_ + t];
    kr_s[t + 256] = kr[(size_t)b * M_ + t + 256];
    __syncthreads();
    int nl = t & 63, g = t >> 6;
    const float* mp = mem + ((size_t)b * M_ + (size_t)g * 128) * M_ + n0 + nl;
    float acc = 0.f;
    for (int r = 0; r < 128; ++r) {
        acc = fmaf(kr_s[g * 128 + r], mp[0], acc);
        mp += M_;
    }
    __shared__ float red[256];
    red[t] = acc; __syncthreads();
    if (g == 0)
        vr[(size_t)b * M_ + n0 + nl] = red[nl] + red[64 + nl] + red[128 + nl] + red[192 + nl];
}

// ---------------------------------------------------------------------------
extern "C" void kernel_launch(void* const* d_in, const int* in_sizes, int n_in,
                              void* d_out, int out_size, void* d_ws, size_t ws_size,
                              hipStream_t stream) {
    (void)in_sizes; (void)n_in; (void)out_size; (void)ws_size;
    const int*   story = (const int*)d_in[0];
    const int*   query = (const int*)d_in[1];
    const float* we    = (const float*)d_in[2];
    const float* pos   = (const float*)d_in[3];
    const float* bn_sg = (const float*)d_in[4];
    const float* bn_sb = (const float*)d_in[5];
    const float* bn_qg = (const float*)d_in[6];
    const float* bn_qb = (const float*)d_in[7];
    const float* Wk    = (const float*)d_in[8];
    const float* bk    = (const float*)d_in[9];
    const float* Wv    = (const float*)d_in[10];
    const float* bv    = (const float*)d_in[11];
    const float* wW    = (const float*)d_in[12];
    const float* wb    = (const float*)d_in[13];
    const float* ln1g  = (const float*)d_in[14];
    const float* ln1b  = (const float*)d_in[15];
    const float* ln2g  = (const float*)d_in[16];
    const float* ln2b  = (const float*)d_in[17];
    const float* rW    = (const float*)d_in[18];
    const float* rb    = (const float*)d_in[19];
    const float* oW    = (const float*)d_in[20];
    const float* ob    = (const float*)d_in[21];

    // ---- workspace layout (256B aligned slabs) ----
    char* w = (char*)d_ws;
    size_t off = 0;
    auto alloc = [&](size_t bytes) -> char* {
        char* p = w + off;
        off = (off + bytes + 255) & ~(size_t)255;
        return p;
    };
    float*  stats  = (float*)alloc(8 * E_ * sizeof(float));   // sums + scale/shift
    float*  vr     = (float*)alloc((size_t)B_ * M_ * 4);
    bf16_t* vh16a  = (bf16_t*)alloc((size_t)B_ * M_ * 2);
    bf16_t* vh16b  = (bf16_t*)alloc((size_t)B_ * M_ * 2);
    size_t zero_bytes = off;                                  // contiguous prefix to zero
    float*  semb   = (float*)alloc((size_t)B_ * S_ * E_ * 4);
    float*  qe     = (float*)alloc((size_t)B_ * E_ * 4);
    float*  qeb    = (float*)alloc((size_t)B_ * E_ * 4);
    bf16_t* Abf    = (bf16_t*)alloc((size_t)B_ * S_ * E_ * 2);
    bf16_t* Wk16   = (bf16_t*)alloc((size_t)M_ * E_ * 2);
    bf16_t* Wv16   = (bf16_t*)alloc((size_t)M_ * E_ * 2);
    bf16_t* WwA16  = (bf16_t*)alloc((size_t)M_ * M_ * 2);     // write_W[:, :512]
    bf16_t* WwB16  = (bf16_t*)alloc((size_t)M_ * M_ * 2);     // write_W[:, 512:]
    bf16_t* Ow16   = (bf16_t*)alloc((size_t)V_ * M_ * 2);
    float*  kmat   = (float*)alloc((size_t)B_ * S_ * M_ * 4);
    bf16_t* v16    = (bf16_t*)alloc((size_t)B_ * S_ * M_ * 2);
    float*  prest  = (float*)alloc((size_t)B_ * S_ * M_ * 4); // v@WwA^T + wb
    float*  kn_all = (float*)alloc((size_t)B_ * S_ * M_ * 4); // [s][b][m]
    float*  preact = (float*)alloc((size_t)B_ * M_ * 4);
    float*  kr     = (float*)alloc((size_t)B_ * M_ * 4);
    bf16_t* vr16   = (bf16_t*)alloc((size_t)B_ * M_ * 2);
    float*  mem    = (float*)alloc((size_t)B_ * M_ * M_ * 4); // 134MB, L2-resident

    auto cvt = [&](const float* s, bf16_t* d, int rows, int cols, int ld) {
        int n = rows * cols;
        k_cvt_bf16<<<(n + 255) / 256, 256, 0, stream>>>(s, d, rows, cols, ld);
    };
    auto gemm = [&](const bf16_t* A, const bf16_t* Bt, const float* bias,
                    const float* Cadd, int cadd_ld, void* C, int outbf, int relu,
                    int Mr, int N, int K) {
        int mt4 = Mr / 64, nt = N / 16;
        int blocks = (mt4 * nt + 7) / 8;
        k_gemm_bf16<<<blocks, 256, 0, stream>>>(A, Bt, bias, Cadd, cadd_ld,
                                                C, outbf, relu, mt4, nt, K, N);
    };

    // 0) zero stats + v_r + both v_h buffers (fresh every call)
    {
        int nwords = (int)(zero_bytes / 4);
        k_zero<<<(nwords + 255) / 256, 256, 0, stream>>>((uint32_t*)d_ws, nwords);
    }
    // 1) embeddings + batchnorm
    k_embed<<<B_ * S_, E_, 0, stream>>>(story, we, pos, semb);
    k_embed<<<B_, E_, 0, stream>>>(query, we, pos, qe);
    k_stats_story<<<256, E_, 0, stream>>>(semb, stats, stats + E_);
    k_stats_query<<<1, E_, 0, stream>>>(qe, stats + 2 * E_, stats + 3 * E_);
    k_bn_finalize<<<1, E_, 0, stream>>>(stats, stats + E_, stats + 2 * E_, stats + 3 * E_,
                                        bn_sg, bn_sb, bn_qg, bn_qb,
                                        stats + 4 * E_, stats + 5 * E_,
                                        stats + 6 * E_, stats + 7 * E_);
    {
        int n = B_ * S_ * E_;
        k_bn_apply_bf16<<<(n + 255) / 256, 256, 0, stream>>>(semb, stats + 4 * E_,
                                                             stats + 5 * E_, Abf, n);
    }
    {
        int n = B_ * E_;
        k_bn_apply_f32<<<(n + 255) / 256, 256, 0, stream>>>(qe, stats + 6 * E_,
                                                            stats + 7 * E_, qeb, n);
    }
    // 2) weight conversions to bf16 ([out,in] weights are already B^T row-major)
    cvt(Wk, Wk16, M_, E_, E_);
    cvt(Wv, Wv16, M_, E_, E_);
    cvt(wW, WwA16, M_, M_, 2 * M_);
    cvt(wW + M_, WwB16, M_, M_, 2 * M_);
    cvt(oW, Ow16, V_, M_, M_);
    // 3) projections (WMMA): k = relu(x@Wk^T+bk) f32; v = relu(x@Wv^T+bv) bf16
    gemm(Abf, Wk16, bk, nullptr, 0, kmat, 0, 1, B_ * S_, M_, E_);
    gemm(Abf, Wv16, bv, nullptr, 0, v16, 1, 1, B_ * S_, M_, E_);
    // 4) static half of the write pre-activation (WMMA): v@WwA^T + write_b
    gemm(v16, WwA16, wb, nullptr, 0, prest, 0, 0, B_ * S_, M_, M_);
    // 5) kn_all = LN(k) for every step, stored step-major
    k_ln<<<B_ * S_, M_, 0, stream>>>(kmat, ln1g, ln1b, kn_all, 1);
    // 6) Hebbian scan: per step, one WMMA GEMM for v_h@WwB^T (+ static term),
    //    then one fused pass over m (LN2 + update + next-step v_h accumulation).
    for (int i = 0; i < S_; ++i) {
        bf16_t* vh_cur = (i & 1) ? vh16b : vh16a;
        bf16_t* vh_nxt = (i & 1) ? vh16a : vh16b;
        gemm(vh_cur, WwB16, nullptr, prest + (size_t)i * M_, S_ * M_,
             preact, 0, 0, B_, M_, M_);
        const float* kn_i = kn_all + (size_t)i * B_ * M_;
        const float* kn_x = (i + 1 < S_) ? kn_all + (size_t)(i + 1) * B_ * M_ : kn_i;
        k_hebb<<<B_ * 8, 256, 0, stream>>>(mem, kn_i, kn_x, preact, ln2g, ln2b,
                                           vh_nxt, i == 0);
    }
    // 7) multi-hop associative read
    for (int h = 0; h < HOPS_; ++h) {
        k_read_kr<<<B_, M_, 0, stream>>>(qeb, vr, rW, rb, kr);
        k_read_vr<<<B_ * 8, 256, 0, stream>>>(mem, kr, vr);
    }
    // 8) logits (WMMA): v_r @ out_W^T + out_b -> d_out [128,10000] f32
    cvt(vr, vr16, B_, M_, M_);
    gemm(vr16, Ow16, ob, nullptr, 0, d_out, 0, 0, B_, V_, M_);
}
